// MyLinearLayer_69123203662202
// MI455X (gfx1250) — compile-verified, matrix operand order checked by reading, not run on hardware
//
#include <hip/hip_runtime.h>

typedef __attribute__((ext_vector_type(16))) _Float16 v16h;
typedef __attribute__((ext_vector_type(8)))  _Float16 v8h;
typedef __attribute__((ext_vector_type(4)))  _Float16 v4h;
typedef __attribute__((ext_vector_type(8)))  float    v8f;

constexpr int BQ   = 32768;  // batch rows
constexpr int IND  = 512;    // inner dim
constexpr int OUTD = 512;    // output dim

constexpr int BM = 128;      // block tile M
constexpr int BN = 128;      // block tile N
constexpr int BK = 32;       // K per LDS stage (one WMMA k-step)
constexpr int LDA = 40;      // padded LDS row pitch in halves (bank-conflict-free b128 frag loads)
constexpr int NSTAGE = IND / BK;          // 16
constexpr unsigned ABUF_BYTES = BM * LDA * 2;   // one A buffer
constexpr unsigned BBUF_BYTES = BN * LDA * 2;   // one B buffer

// trunc-toward-zero fixed-point quantize, then keep top-4 significant bits of |.|
__device__ __forceinline__ _Float16 qtop4(float v, float s) {
    int a = (int)(v * s);                       // C cast == jnp astype(int32): trunc toward zero
    unsigned m = (a < 0) ? (unsigned)(-a) : (unsigned)a;
    if (m) {
        int sh = (32 - __clz((int)m)) - 4;      // bit_length - 4
        if (sh > 0) m = (m >> sh) << sh;
    }
    float r = (float)m;
    return (_Float16)((a < 0) ? -r : r);        // <=4 significant bits: exact in f16
}

__global__ __launch_bounds__(256)
void quant_kernel(const float* __restrict__ in, _Float16* __restrict__ out,
                  int n, const int* __restrict__ sf_p) {
    const float s = (float)(1 << *sf_p);
    int i = (blockIdx.x * blockDim.x + threadIdx.x) * 4;
    if (i + 3 >= n) {
        if (i >= n) return;
        for (; i < n; ++i) out[i] = qtop4(in[i], s);
        return;
    }
    float4 v = *(const float4*)(in + i);
    v4h h;
    h[0] = qtop4(v.x, s);
    h[1] = qtop4(v.y, s);
    h[2] = qtop4(v.z, s);
    h[3] = qtop4(v.w, s);
    *(v4h*)(out + i) = h;
}

// CDNA5 async memory->LDS copy, 16B per lane, tracked by ASYNCcnt.
// vdst = per-lane LDS byte offset (wave-relative), vaddr = per-lane 64-bit global addr.
__device__ __forceinline__ void async_cp_b128(unsigned lds_off, const void* gaddr) {
    asm volatile("global_load_async_to_lds_b128 %0, %1, off"
                 :: "v"(lds_off), "v"((unsigned long long)(uintptr_t)gaddr)
                 : "memory");
}

__global__ __launch_bounds__(256)
void gemm_wmma_kernel(const _Float16* __restrict__ A,   // tx [BQ][IND]
                      const _Float16* __restrict__ Bw,  // tw [OUTD][IND]
                      const float* __restrict__ bias,   // [OUTD]
                      const int* __restrict__ wsf_p,
                      const int* __restrict__ asf_p,
                      float* __restrict__ C) {          // [BQ][OUTD]
    __shared__ _Float16 sA[2][BM * LDA];
    __shared__ _Float16 sB[2][BN * LDA];

    const int t    = threadIdx.x;
    const int lane = t & 31;
    const int wave = t >> 5;
    const int m16  = lane & 15;   // row/col within 16x16 tile
    const int hi   = lane >> 4;   // half-wave select

    const int bm = blockIdx.y * BM;
    const int bn = blockIdx.x * BN;

    const int wm = (wave & 3) * 32;    // wave M offset inside block tile (4 waves in M)
    const int wn = (wave >> 2) * 64;   // wave N offset inside block tile (2 waves in N)

    // Global->LDS staging map: 512 chunks of 8 halves per operand tile, 2 per thread.
    const int r0 = t >> 2;                 // rows 0..63  (chunk t); chunk t+256 -> rows 64..127
    const int c0 = (t & 3) * 8;            // col 0/8/16/24 within BK
    const _Float16* gA0 = A  + (size_t)(bm + r0)      * IND + c0;
    const _Float16* gA1 = A  + (size_t)(bm + r0 + 64) * IND + c0;
    const _Float16* gB0 = Bw + (size_t)(bn + r0)      * IND + c0;
    const _Float16* gB1 = Bw + (size_t)(bn + r0 + 64) * IND + c0;

    // Wave-relative LDS byte offsets (low 32 bits of the flat shared pointer).
    const unsigned oA0 = (unsigned)(uintptr_t)(&sA[0][r0 * LDA + c0]);
    const unsigned oA1 = (unsigned)(uintptr_t)(&sA[0][(r0 + 64) * LDA + c0]);
    const unsigned oB0 = (unsigned)(uintptr_t)(&sB[0][r0 * LDA + c0]);
    const unsigned oB1 = (unsigned)(uintptr_t)(&sB[0][(r0 + 64) * LDA + c0]);

    auto issue_stage = [&](int stage, int buf) {
        const int koff = stage * BK;
        const unsigned da = buf ? ABUF_BYTES : 0u;
        const unsigned db = buf ? BBUF_BYTES : 0u;
        async_cp_b128(oA0 + da, gA0 + koff);
        async_cp_b128(oA1 + da, gA1 + koff);
        async_cp_b128(oB0 + db, gB0 + koff);
        async_cp_b128(oB1 + db, gB1 + koff);
    };

    v8f acc[2][4] = {};

    // Double-buffered async pipeline: prime two stages.
    issue_stage(0, 0);
    issue_stage(1, 1);

    for (int i = 0; i < NSTAGE; ++i) {
        // Stage i complete (4 async ops of stage i+1 may remain in flight).
        if (i + 1 < NSTAGE)
            asm volatile("s_wait_asynccnt 4" ::: "memory");
        else
            asm volatile("s_wait_asynccnt 0" ::: "memory");
        __syncthreads();                      // all waves' stage-i data visible

        const _Float16* aBase = sA[i & 1];
        const _Float16* bBase = sB[i & 1];

        // Per-lane 16-bit A-matrix fragment layout (ISA 7.12.2):
        // lane L: m = L&15; halves 0..7  = K[hi*8 .. +8), halves 8..15 = K[16+hi*8 .. +8)
        v16h af[2];
        #pragma unroll
        for (int mt = 0; mt < 2; ++mt) {
            const _Float16* p = aBase + (wm + mt * 16 + m16) * LDA + hi * 8;
            v8h lo = *(const v8h*)(p);
            v8h hh = *(const v8h*)(p + 16);
            af[mt] = __builtin_shufflevector(lo, hh,
                     0,1,2,3,4,5,6,7,8,9,10,11,12,13,14,15);
        }
        v16h bf[4];
        #pragma unroll
        for (int nt = 0; nt < 4; ++nt) {
            const _Float16* p = bBase + (wn + nt * 16 + m16) * LDA + hi * 8;
            v8h lo = *(const v8h*)(p);
            v8h hh = *(const v8h*)(p + 16);
            bf[nt] = __builtin_shufflevector(lo, hh,
                     0,1,2,3,4,5,6,7,8,9,10,11,12,13,14,15);
        }

        #pragma unroll
        for (int mt = 0; mt < 2; ++mt)
            #pragma unroll
            for (int nt = 0; nt < 4; ++nt)
                acc[mt][nt] = __builtin_amdgcn_wmma_f32_16x16x32_f16(
                    false, af[mt], false, bf[nt],
                    (short)0, acc[mt][nt], false, false);

        __syncthreads();                      // buffer i&1 fully consumed by all waves
        if (i + 2 < NSTAGE)
            issue_stage(i + 2, i & 1);        // refill the freed buffer
    }

    // Epilogue: C = acc * 2^-(wsf+asf) + bias   (exact power-of-two scale)
    const float scale = ldexpf(1.0f, -(*wsf_p + *asf_p));
    #pragma unroll
    for (int nt = 0; nt < 4; ++nt) {
        const int gn = bn + wn + nt * 16 + m16;
        const float bv = bias[gn];
        #pragma unroll
        for (int mt = 0; mt < 2; ++mt) {
            #pragma unroll
            for (int r = 0; r < 8; ++r) {
                // C/D layout: VGPR r, lanes<16 -> M=r, lanes>=16 -> M=r+8; N = lane&15
                const int gm = bm + wm + mt * 16 + r + 8 * hi;
                C[(size_t)gm * OUTD + gn] = acc[mt][nt][r] * scale + bv;
            }
        }
    }
}

extern "C" void kernel_launch(void* const* d_in, const int* in_sizes, int n_in,
                              void* d_out, int out_size, void* d_ws, size_t ws_size,
                              hipStream_t stream) {
    const float* x    = (const float*)d_in[0];   // [BQ, IND]
    const float* w    = (const float*)d_in[1];   // [OUTD, IND]
    const float* bias = (const float*)d_in[2];   // [OUTD]
    const int*   wsf  = (const int*)d_in[3];     // scalar
    const int*   asf  = (const int*)d_in[4];     // scalar

    _Float16* tw = (_Float16*)d_ws;                       // 512*512*2   = 512 KB
    _Float16* tx = tw + (size_t)OUTD * IND;               // 32768*512*2 = 32 MB
    float*    out = (float*)d_out;

    // quantize weight and activations to exact-f16 top-4-bit values
    quant_kernel<<<(OUTD * IND) / 1024, 256, 0, stream>>>(w, tw, OUTD * IND, wsf);
    quant_kernel<<<(BQ * IND) / 1024, 256, 0, stream>>>(x, tx, BQ * IND, asf);

    dim3 grid(OUTD / BN, BQ / BM);
    gemm_wmma_kernel<<<grid, 256, 0, stream>>>(tx, tw, bias, wsf, asf, out);
}
